// Attn_86741159510399
// MI455X (gfx1250) — compile-verified, matrix operand order checked by reading, not run on hardware
//
#include <hip/hip_runtime.h>
#include <hip/hip_bf16.h>

// ---------------------------------------------------------------------------
// Fused attention for B=8, S=2048, D=1024 (fp32 in/out), bf16 WMMA internally.
//
//   K1 qkv_proj : Q/K/V = x@W^T+b  (f32 staged->bf16 in LDS; Q,K row-major
//                 bf16; V stored transposed so PV GEMM is K-contiguous)
//   K2 scores   : S = (Q@K^T)/32, mask==0 -> -1e9, bf16  (async-LDS staged)
//   K3 softmax  : in-place row softmax (wave per row, wave32 shuffles)
//   K4 pv       : O = P@Vt^T -> f32 output                (async-LDS staged)
// ---------------------------------------------------------------------------

typedef __attribute__((ext_vector_type(16))) __bf16 v16bf;
typedef __attribute__((ext_vector_type(8)))  __bf16 v8bf;
typedef __attribute__((ext_vector_type(8)))  float  v8f;

#define LDA 48           // LDS row pitch in bf16 elems (96B: 16B-aligned)
#define TM 128           // workgroup M tile
#define TN 128           // workgroup N tile
#define TK 32            // K step (one bf16 WMMA)

union FragU { v16bf v; v8bf h[2]; };
union AccU  { v8f v; float f[8]; };

// One 16x32 bf16 fragment (ISA 7.12.2 layout): lane%16 = row (A) / col (B),
// lane/16 selects K-half; two 16-byte LDS reads per lane.
__device__ __forceinline__ v16bf ld_frag(const __bf16* __restrict__ base, int lane) {
  const int r = lane & 15;
  const int h = lane >> 4;
  const __bf16* p = base + r * LDA + h * 8;
  FragU f;
  f.h[0] = *reinterpret_cast<const v8bf*>(p);
  f.h[1] = *reinterpret_cast<const v8bf*>(p + 16);
  return f.v;
}

__device__ __forceinline__ v8f wmma_bf16(v16bf a, v16bf b, v8f c) {
  return __builtin_amdgcn_wmma_f32_16x16x32_bf16(
      false, a, false, b, (short)0, c, false, false);
}

// gfx1250 async copy: global -> LDS, 16B per enabled lane, ASYNCcnt-tracked.
__device__ __forceinline__ void async_ld_b128(const __bf16* lds_dst,
                                              const __bf16* gsrc) {
  const unsigned lds_off = (unsigned)(uintptr_t)lds_dst;  // flat LDS addr[31:0]
  asm volatile("global_load_async_to_lds_b128 %0, %1, off"
               :: "v"(lds_off), "v"((unsigned long long)(uintptr_t)gsrc)
               : "memory");
}
__device__ __forceinline__ void wait_async0() {
  asm volatile("s_wait_asynccnt 0" ::: "memory");
}

// ---------------------------------------------------------------------------
// K1: fused QKV projection.  grid = (16384/TM, 1024/TN, 3), block 256 (8 waves)
// ---------------------------------------------------------------------------
__global__ __launch_bounds__(256)
void qkv_proj_kernel(const float* __restrict__ x,
                     const float* __restrict__ Wq, const float* __restrict__ Wk,
                     const float* __restrict__ Wv,
                     const float* __restrict__ bq, const float* __restrict__ bk,
                     const float* __restrict__ bv,
                     __bf16* __restrict__ Qb, __bf16* __restrict__ Kb,
                     __bf16* __restrict__ Vt) {
  __shared__ __align__(16) __bf16 As[TM * LDA];
  __shared__ __align__(16) __bf16 Bs[TN * LDA];

  const int z = blockIdx.z;                      // 0:Q 1:K 2:V
  const float* W    = (z == 0) ? Wq : (z == 1) ? Wk : Wv;
  const float* bias = (z == 0) ? bq : (z == 1) ? bk : bv;

  const int lane = threadIdx.x & 31;
  const int wave = threadIdx.x >> 5;
  const int wm = wave & 3;                       // 4 waves along M (32 rows each)
  const int wn = wave >> 2;                      // 2 waves along N (64 cols each)
  const size_t row0 = (size_t)blockIdx.x * TM;
  const int    col0 = blockIdx.y * TN;

  const float* Aptr = x + row0 * 1024;
  const float* Bptr = W + (size_t)col0 * 1024;

  v8f C[2][4] = {};

  for (int k0 = 0; k0 < 1024; k0 += TK) {
    __syncthreads();
    for (int u = threadIdx.x; u < TM * 8; u += 256) {   // 128x32 f32 -> bf16
      const int r = u >> 3, q = u & 7;
      const float4 t = *reinterpret_cast<const float4*>(Aptr + (size_t)r * 1024 + k0 + q * 4);
      __bf16* d = &As[r * LDA + q * 4];
      d[0] = (__bf16)t.x; d[1] = (__bf16)t.y; d[2] = (__bf16)t.z; d[3] = (__bf16)t.w;
    }
    for (int u = threadIdx.x; u < TN * 8; u += 256) {   // 128x32 f32 -> bf16
      const int r = u >> 3, q = u & 7;
      const float4 t = *reinterpret_cast<const float4*>(Bptr + (size_t)r * 1024 + k0 + q * 4);
      __bf16* d = &Bs[r * LDA + q * 4];
      d[0] = (__bf16)t.x; d[1] = (__bf16)t.y; d[2] = (__bf16)t.z; d[3] = (__bf16)t.w;
    }
    if (k0 + TK < 1024)
      __builtin_prefetch(Aptr + (size_t)(threadIdx.x >> 1) * 1024 + k0 + TK, 0, 1);
    __syncthreads();

    const v16bf a0 = ld_frag(&As[(wm * 32 + 0)  * LDA], lane);
    const v16bf a1 = ld_frag(&As[(wm * 32 + 16) * LDA], lane);
    #pragma unroll
    for (int fn = 0; fn < 4; ++fn) {
      const v16bf b = ld_frag(&Bs[(wn * 64 + fn * 16) * LDA], lane);
      C[0][fn] = wmma_bf16(a0, b, C[0][fn]);
      C[1][fn] = wmma_bf16(a1, b, C[1][fn]);
    }
  }

  __bf16* outQK = (z == 0) ? Qb : Kb;
  for (int fm = 0; fm < 2; ++fm)
    for (int fn = 0; fn < 4; ++fn) {
      AccU acc; acc.v = C[fm][fn];
      const int e = col0 + wn * 64 + fn * 16 + (lane & 15);
      const float bv_ = bias[e];
      for (int v = 0; v < 8; ++v) {
        const int ml = wm * 32 + fm * 16 + (lane >> 4) * 8 + v;
        const size_t r = row0 + ml;
        const float val = acc.f[v] + bv_;
        if (z < 2) {
          outQK[r * 1024 + e] = (__bf16)val;
        } else {                                  // V stored transposed per batch
          const size_t b = r >> 11, s = r & 2047;
          Vt[((b * 1024) + e) * 2048 + s] = (__bf16)val;
        }
      }
    }
}

// ---------------------------------------------------------------------------
// Shared async double-buffered bf16 GEMM body (used by K2 and K4).
// ---------------------------------------------------------------------------
#define STAGE_ASYNC(buf, k0)                                                   \
  do {                                                                         \
    for (int u = threadIdx.x; u < TM * 4; u += 256) {                          \
      const int r = u >> 2, q = u & 3;                                         \
      async_ld_b128(&As[buf][r * LDA + q * 8],                                 \
                    Aptr + (size_t)r * strideA + (k0) + q * 8);                \
    }                                                                          \
    for (int u = threadIdx.x; u < TN * 4; u += 256) {                          \
      const int r = u >> 2, q = u & 3;                                         \
      async_ld_b128(&Bs[buf][r * LDA + q * 8],                                 \
                    Bptr + (size_t)r * strideB + (k0) + q * 8);                \
    }                                                                          \
  } while (0)

// ---------------------------------------------------------------------------
// K2: scores = (Q@K^T)*scale with mask.  grid = (16384/TM, 2048/TN), block 256
// ---------------------------------------------------------------------------
__global__ __launch_bounds__(256)
void scores_kernel(const __bf16* __restrict__ Qb, const __bf16* __restrict__ Kb,
                   const int* __restrict__ mask, __bf16* __restrict__ Sc) {
  __shared__ __align__(16) __bf16 As[2][TM * LDA];
  __shared__ __align__(16) __bf16 Bs[2][TN * LDA];

  const int lane = threadIdx.x & 31;
  const int wave = threadIdx.x >> 5;
  const int wm = wave & 3, wn = wave >> 2;

  const size_t row0 = (size_t)blockIdx.x * TM;    // global query row (b*2048+q)
  const int b = (int)(row0 >> 11);
  const int key0 = blockIdx.y * TN;

  const __bf16* Aptr = Qb + row0 * 1024;
  const __bf16* Bptr = Kb + ((size_t)b * 2048 + key0) * 1024;
  const int strideA = 1024, strideB = 1024;

  v8f C[2][4] = {};

  STAGE_ASYNC(0, 0);
  wait_async0();
  __syncthreads();

  int cur = 0;
  for (int k0 = 0; k0 < 1024; k0 += TK) {
    const int nxt = cur ^ 1;
    if (k0 + TK < 1024) STAGE_ASYNC(nxt, k0 + TK);

    const v16bf a0 = ld_frag(&As[cur][(wm * 32 + 0)  * LDA], lane);
    const v16bf a1 = ld_frag(&As[cur][(wm * 32 + 16) * LDA], lane);
    #pragma unroll
    for (int fn = 0; fn < 4; ++fn) {
      const v16bf bb = ld_frag(&Bs[cur][(wn * 64 + fn * 16) * LDA], lane);
      C[0][fn] = wmma_bf16(a0, bb, C[0][fn]);
      C[1][fn] = wmma_bf16(a1, bb, C[1][fn]);
    }

    wait_async0();
    __syncthreads();
    cur = nxt;
  }

  const float scale = 0.03125f;                   // 1/sqrt(1024)
  for (int fm = 0; fm < 2; ++fm)
    for (int fn = 0; fn < 4; ++fn) {
      AccU acc; acc.v = C[fm][fn];
      const int nk = key0 + wn * 64 + fn * 16 + (lane & 15);
      const int mk = mask[b * 2048 + nk];
      for (int v = 0; v < 8; ++v) {
        const int ml = wm * 32 + fm * 16 + (lane >> 4) * 8 + v;
        const size_t r = row0 + ml;
        const float s = mk ? acc.f[v] * scale : -1.0e9f;
        Sc[r * 2048 + nk] = (__bf16)s;
      }
    }
}

// ---------------------------------------------------------------------------
// K3: in-place row softmax over 2048 elems. One wave per row; 8 rows / block.
// ---------------------------------------------------------------------------
__global__ __launch_bounds__(256)
void softmax_kernel(__bf16* __restrict__ Sc) {
  const int lane = threadIdx.x & 31;
  const int wave = threadIdx.x >> 5;
  const size_t row = (size_t)blockIdx.x * 8 + wave;
  __bf16* p = Sc + row * 2048;

  float vals[64];
  float m = -3.4e38f;
  for (int i = 0; i < 64; ++i) {
    vals[i] = (float)p[lane + i * 32];
    m = fmaxf(m, vals[i]);
  }
  for (int off = 16; off > 0; off >>= 1) m = fmaxf(m, __shfl_xor(m, off, 32));

  float s = 0.0f;
  for (int i = 0; i < 64; ++i) { vals[i] = __expf(vals[i] - m); s += vals[i]; }
  for (int off = 16; off > 0; off >>= 1) s += __shfl_xor(s, off, 32);

  const float inv = 1.0f / s;
  for (int i = 0; i < 64; ++i) p[lane + i * 32] = (__bf16)(vals[i] * inv);
}

// ---------------------------------------------------------------------------
// K4: O = P @ V  (B = Vt[e][k], K-contiguous).  grid = (16384/TM, 1024/TN).
// ---------------------------------------------------------------------------
__global__ __launch_bounds__(256)
void pv_kernel(const __bf16* __restrict__ P, const __bf16* __restrict__ Vt,
               float* __restrict__ O) {
  __shared__ __align__(16) __bf16 As[2][TM * LDA];
  __shared__ __align__(16) __bf16 Bs[2][TN * LDA];

  const int lane = threadIdx.x & 31;
  const int wave = threadIdx.x >> 5;
  const int wm = wave & 3, wn = wave >> 2;

  const size_t row0 = (size_t)blockIdx.x * TM;    // global query row (b*2048+q)
  const int b = (int)(row0 >> 11);
  const int e0 = blockIdx.y * TN;

  const __bf16* Aptr = P + row0 * 2048;
  const __bf16* Bptr = Vt + ((size_t)b * 1024 + e0) * 2048;
  const int strideA = 2048, strideB = 2048;

  v8f C[2][4] = {};

  STAGE_ASYNC(0, 0);
  wait_async0();
  __syncthreads();

  int cur = 0;
  for (int k0 = 0; k0 < 2048; k0 += TK) {
    const int nxt = cur ^ 1;
    if (k0 + TK < 2048) STAGE_ASYNC(nxt, k0 + TK);

    const v16bf a0 = ld_frag(&As[cur][(wm * 32 + 0)  * LDA], lane);
    const v16bf a1 = ld_frag(&As[cur][(wm * 32 + 16) * LDA], lane);
    #pragma unroll
    for (int fn = 0; fn < 4; ++fn) {
      const v16bf bb = ld_frag(&Bs[cur][(wn * 64 + fn * 16) * LDA], lane);
      C[0][fn] = wmma_bf16(a0, bb, C[0][fn]);
      C[1][fn] = wmma_bf16(a1, bb, C[1][fn]);
    }

    wait_async0();
    __syncthreads();
    cur = nxt;
  }

  for (int fm = 0; fm < 2; ++fm)
    for (int fn = 0; fn < 4; ++fn) {
      AccU acc; acc.v = C[fm][fn];
      const int e = e0 + wn * 64 + fn * 16 + (lane & 15);
      for (int v = 0; v < 8; ++v) {
        const int ml = wm * 32 + fm * 16 + (lane >> 4) * 8 + v;
        const size_t r = row0 + ml;
        O[r * 1024 + e] = acc.f[v];
      }
    }
}

// ---------------------------------------------------------------------------
extern "C" void kernel_launch(void* const* d_in, const int* in_sizes, int n_in,
                              void* d_out, int out_size, void* d_ws, size_t ws_size,
                              hipStream_t stream) {
  const float* x    = (const float*)d_in[0];   // [8,2048,1024]
  const int*   mask = (const int*)  d_in[1];   // [8,2048]
  const float* Wq   = (const float*)d_in[2];
  const float* bq   = (const float*)d_in[3];
  const float* Wk   = (const float*)d_in[4];
  const float* bk   = (const float*)d_in[5];
  const float* Wv   = (const float*)d_in[6];
  const float* bv   = (const float*)d_in[7];
  float* O = (float*)d_out;                    // [8,2048,1024] f32

  const size_t QKV_ELEMS = (size_t)8 * 2048 * 1024;
  __bf16* Qb = (__bf16*)d_ws;
  __bf16* Kb = Qb + QKV_ELEMS;
  __bf16* Vt = Kb + QKV_ELEMS;
  __bf16* Sc = Vt + QKV_ELEMS;                 // [16384][2048] bf16

  const dim3 blk(256);
  qkv_proj_kernel<<<dim3(16384 / TM, 1024 / TN, 3), blk, 0, stream>>>(
      x, Wq, Wk, Wv, bq, bk, bv, Qb, Kb, Vt);
  scores_kernel<<<dim3(16384 / TM, 2048 / TN), blk, 0, stream>>>(Qb, Kb, mask, Sc);
  softmax_kernel<<<dim3(16384 / 8), blk, 0, stream>>>(Sc);
  pv_kernel<<<dim3(16384 / TM, 1024 / TN), blk, 0, stream>>>(Sc, Vt, O);
}
